// GNN_Model_6493990552142
// MI455X (gfx1250) — matrix operand order repaired and checked by reading, not compiled, not run on hardware
//
#include <hip/hip_runtime.h>

// ---------------- problem constants (static in the reference) ----------------
#define N_LINKS 10000
#define N_PATHS 50000
#define PLEN    5
#define EDG     (N_PATHS * PLEN)
#define T_ITERS 8
// D = H = 32, gates = 96, RU = 256

typedef _Float16 h4   __attribute__((ext_vector_type(4)));
typedef _Float16 h8   __attribute__((ext_vector_type(8)));
typedef _Float16 v16h __attribute__((ext_vector_type(16)));
typedef float    v8f  __attribute__((ext_vector_type(8)));
typedef float    f4   __attribute__((ext_vector_type(4)));

// ---------------- WMMA + fragment helpers ----------------
__device__ __forceinline__ v8f wmma_f16(v16h a, v16h b, v8f c) {
  // D = A(16x32 f16) * B(32x16 f16) + C(16x16 f32)
  return __builtin_amdgcn_wmma_f32_16x16x32_f16(false, a, false, b, (short)0, c,
                                                false, false);
}

__device__ __forceinline__ v8f splat8(float x) {
  v8f r;
#pragma unroll
  for (int i = 0; i < 8; ++i) r[i] = x;
  return r;
}

// B fragment, pre-packed: 512 halfs per fragment, lane-major, 16 halfs/lane.
__device__ __forceinline__ v16h load_b_frag(const _Float16* __restrict__ base,
                                            int lane) {
  union { v16h v; h8 h[2]; } u;
  const h8* p = (const h8*)(base + lane * 16);
  u.h[0] = p[0];
  u.h[1] = p[1];
  return u.v;
}

// A fragment from a row-major f32 global row of 32 floats.
// element e : K = hi*16 + e
__device__ __forceinline__ v16h load_a_global_f32(const float* __restrict__ row,
                                                  int hi) {
  const f4* p0 = (const f4*)(row + hi * 16);
  const f4* p1 = (const f4*)(row + hi * 16 + 8);
  f4 a = p0[0], b = p0[1], c = p1[0], d = p1[1];
  v16h r;
#pragma unroll
  for (int i = 0; i < 4; ++i) {
    r[i]      = (_Float16)a[i];
    r[4 + i]  = (_Float16)b[i];
    r[8 + i]  = (_Float16)c[i];
    r[12 + i] = (_Float16)d[i];
  }
  return r;
}

// A fragment from row-major f16 LDS tile (row stride in halfs).
__device__ __forceinline__ v16h load_a_lds(const _Float16* sh, int lo, int hi,
                                           int stride) {
  const _Float16* r = sh + lo * stride;
  union { v16h v; h8 h[2]; } u;
  u.h[0] = *(const h8*)(r + hi * 16);
  u.h[1] = *(const h8*)(r + hi * 16 + 8);
  return u.v;
}

__device__ __forceinline__ float sigm_(float x) {
  return 1.0f / (1.0f + __expf(-x));
}
__device__ __forceinline__ float tanh_(float x) {
  return 1.0f - 2.0f / (1.0f + __expf(2.0f * x));
}

// One GRU step for a 16-row tile held by one wave.
// xa : A-frag of input x (16x32), hA : A-frag of hidden, hD : hidden in C/D layout.
__device__ __forceinline__ void gru_step(v16h xa, v16h hA, float hD[2][8],
                                         const _Float16* __restrict__ WiT,
                                         const _Float16* __restrict__ WhT,
                                         const float bi_l[6], const float bh_l[6],
                                         int lane, int lo, int hi,
                                         _Float16* shh, bool writeLds) {
  v8f gi[6], gh[6];
#pragma unroll
  for (int f = 0; f < 6; ++f)
    gi[f] = wmma_f16(xa, load_b_frag(WiT + f * 512, lane), splat8(bi_l[f]));
#pragma unroll
  for (int f = 0; f < 6; ++f)
    gh[f] = wmma_f16(hA, load_b_frag(WhT + f * 512, lane), splat8(bh_l[f]));
#pragma unroll
  for (int t2 = 0; t2 < 2; ++t2) {
#pragma unroll
    for (int v = 0; v < 8; ++v) {
      float r = sigm_(gi[t2][v] + gh[t2][v]);
      float z = sigm_(gi[2 + t2][v] + gh[2 + t2][v]);
      float n = tanh_(gi[4 + t2][v] + r * gh[4 + t2][v]);
      hD[t2][v] = (1.0f - z) * n + z * hD[t2][v];
    }
  }
  if (writeLds) {
#pragma unroll
    for (int t2 = 0; t2 < 2; ++t2)
#pragma unroll
      for (int v = 0; v < 8; ++v)
        shh[(v + 8 * hi) * 32 + t2 * 16 + lo] = (_Float16)hD[t2][v];
  }
}

// ---------------- init / pack kernels ----------------
__global__ void __launch_bounds__(256) init_states_kernel(
    const float* __restrict__ traffic, const float* __restrict__ packets,
    const float* __restrict__ tdp, const float* __restrict__ capacity,
    float* __restrict__ path_state, float* __restrict__ link_state) {
  int i = blockIdx.x * 256 + threadIdx.x;
  if (i < N_LINKS * 32) {
    int r = i >> 5, c = i & 31;
    link_state[i] = (c == 0) ? capacity[r] : 0.0f;
  }
  if (i < N_PATHS * 32) {
    int r = i >> 5, c = i & 31;
    float v = 0.0f;
    if (c == 0) v = traffic[r];
    else if (c == 1) v = packets[r];
    else if (c < 14) v = tdp[r * 12 + (c - 2)];
    path_state[i] = v;
  }
}

// Pack W [outF x inF] row-major f32 into B-fragment-ordered f16:
// frag index = kc*nTilesN + f ; lane = n&15 | (kgroup<<4) ; element e -> K=kgroup*16+e
__global__ void __launch_bounds__(256) pack_weight_kernel(
    const float* __restrict__ W, _Float16* __restrict__ dst, int nTilesN,
    int nKC, int inF) {
  int id = blockIdx.x * 256 + threadIdx.x;
  int total = nTilesN * nKC * 512;
  if (id >= total) return;
  int frag = id >> 9;
  int within = id & 511;
  int lane = within >> 4;
  int e = within & 15;
  int kc = frag / nTilesN;
  int f = frag % nTilesN;
  int n = f * 16 + (lane & 15);
  int k = kc * 32 + (lane >> 4) * 16 + e;
  dst[id] = (_Float16)W[n * inF + k];
}

__global__ void __launch_bounds__(256) zero_kernel(float* __restrict__ p, int n) {
  int i = blockIdx.x * 256 + threadIdx.x;
  if (i < n) p[i] = 0.0f;
}

// ---------------- path GRU: 5 steps over gathered link states ----------------
__global__ void __launch_bounds__(128) path_gru_kernel(
    const float* __restrict__ link_state, float* __restrict__ path_state,
    const int* __restrict__ link_to_path,
    const _Float16* __restrict__ WiT, const _Float16* __restrict__ WhT,
    const float* __restrict__ bi, const float* __restrict__ bh) {
  __shared__ __align__(16) _Float16 smem[4][16 * 32];
  const int wave = threadIdx.x >> 5, lane = threadIdx.x & 31;
  const int lo = lane & 15, hi = lane >> 4;
  const int tile = blockIdx.x * 4 + wave;
  if (tile * 16 >= N_PATHS) return;  // wave-uniform exit (no block barriers used)
  const int row0 = tile * 16;
  _Float16* shh = smem[wave];

  float bi_l[6], bh_l[6];
#pragma unroll
  for (int f = 0; f < 6; ++f) {
    bi_l[f] = bi[f * 16 + lo];
    bh_l[f] = bh[f * 16 + lo];
  }

  // stage initial hidden into LDS (f16, row-major 16x32) for A-frag reads
  const f4* src = (const f4*)(path_state + row0 * 32);
#pragma unroll
  for (int i = 0; i < 4; ++i) {
    int idx = i * 32 + lane;
    f4 v = src[idx];
    h4 hv;
#pragma unroll
    for (int j = 0; j < 4; ++j) hv[j] = (_Float16)v[j];
    *(h4*)(shh + (idx >> 3) * 32 + (idx & 7) * 4) = hv;
  }
  asm volatile("s_wait_dscnt 0" ::: "memory");
  v16h hA = load_a_lds(shh, lo, hi, 32);

  // carried hidden in f32, C/D layout (row = v + 8*hi, col = t2*16 + lo)
  float hD[2][8];
#pragma unroll
  for (int t2 = 0; t2 < 2; ++t2)
#pragma unroll
    for (int v = 0; v < 8; ++v)
      hD[t2][v] = path_state[(row0 + v + 8 * hi) * 32 + t2 * 16 + lo];

  const int p = row0 + lo;  // path index for this lane's A-row
#pragma unroll
  for (int t = 0; t < PLEN; ++t) {
    int li = link_to_path[p * PLEN + t];  // edge (p,t) is at p*PLEN+t
    v16h xa = load_a_global_f32(link_state + li * 32, hi);
    gru_step(xa, hA, hD, WiT, WhT, bi_l, bh_l, lane, lo, hi, shh, t < PLEN - 1);
    if (t < PLEN - 1) {
      asm volatile("s_wait_dscnt 0" ::: "memory");
      hA = load_a_lds(shh, lo, hi, 32);
    }
  }

#pragma unroll
  for (int t2 = 0; t2 < 2; ++t2)
#pragma unroll
    for (int v = 0; v < 8; ++v)
      path_state[(row0 + v + 8 * hi) * 32 + t2 * 16 + lo] = hD[t2][v];
}

// ---------------- scatter: path_sum[link] += path_state[path] ----------------
__global__ void __launch_bounds__(256) scatter_add_kernel(
    const float* __restrict__ path_state, const int* __restrict__ path_to_link,
    const int* __restrict__ sequence_links, float* __restrict__ path_sum) {
  int e = blockIdx.x * 256 + threadIdx.x;
  if (e >= EDG) return;
  int p = path_to_link[e];
  int l = sequence_links[e];
  const float* s = path_state + p * 32;
  float* d = path_sum + l * 32;
#pragma unroll
  for (int k = 0; k < 32; ++k) atomicAdd(d + k, s[k]);
}

// ---------------- link GRU: single step ----------------
__global__ void __launch_bounds__(128) link_gru_kernel(
    float* __restrict__ link_state, const float* __restrict__ path_sum,
    const _Float16* __restrict__ WiT, const _Float16* __restrict__ WhT,
    const float* __restrict__ bi, const float* __restrict__ bh) {
  __shared__ __align__(16) _Float16 smem[4][16 * 32];
  const int wave = threadIdx.x >> 5, lane = threadIdx.x & 31;
  const int lo = lane & 15, hi = lane >> 4;
  const int tile = blockIdx.x * 4 + wave;
  if (tile * 16 >= N_LINKS) return;
  const int row0 = tile * 16;
  _Float16* shh = smem[wave];

  float bi_l[6], bh_l[6];
#pragma unroll
  for (int f = 0; f < 6; ++f) {
    bi_l[f] = bi[f * 16 + lo];
    bh_l[f] = bh[f * 16 + lo];
  }

  const f4* src = (const f4*)(link_state + row0 * 32);
#pragma unroll
  for (int i = 0; i < 4; ++i) {
    int idx = i * 32 + lane;
    f4 v = src[idx];
    h4 hv;
#pragma unroll
    for (int j = 0; j < 4; ++j) hv[j] = (_Float16)v[j];
    *(h4*)(shh + (idx >> 3) * 32 + (idx & 7) * 4) = hv;
  }
  asm volatile("s_wait_dscnt 0" ::: "memory");
  v16h hA = load_a_lds(shh, lo, hi, 32);

  float hD[2][8];
#pragma unroll
  for (int t2 = 0; t2 < 2; ++t2)
#pragma unroll
    for (int v = 0; v < 8; ++v)
      hD[t2][v] = link_state[(row0 + v + 8 * hi) * 32 + t2 * 16 + lo];

  v16h xa = load_a_global_f32(path_sum + (row0 + lo) * 32, hi);
  gru_step(xa, hA, hD, WiT, WhT, bi_l, bh_l, lane, lo, hi, shh, false);

#pragma unroll
  for (int t2 = 0; t2 < 2; ++t2)
#pragma unroll
    for (int v = 0; v < 8; ++v)
      link_state[(row0 + v + 8 * hi) * 32 + t2 * 16 + lo] = hD[t2][v];
}

// ---------------- readout MLP: 32 -> 256 -> 256 -> 1 ----------------
__global__ void __launch_bounds__(128) readout_kernel(
    const float* __restrict__ path_state,
    const _Float16* __restrict__ W1T, const float* __restrict__ b1,
    const _Float16* __restrict__ W2T, const float* __restrict__ b2,
    const float* __restrict__ W3, const float* __restrict__ b3,
    float* __restrict__ out) {
  __shared__ __align__(16) _Float16 smem[4][16 * 256];
  const int wave = threadIdx.x >> 5, lane = threadIdx.x & 31;
  const int lo = lane & 15, hi = lane >> 4;
  const int tile = blockIdx.x * 4 + wave;
  if (tile * 16 >= N_PATHS) return;
  const int row0 = tile * 16;
  _Float16* shh = smem[wave];

  // stage path_state tile (16x32) into LDS f16, build A fragment
  const f4* src = (const f4*)(path_state + row0 * 32);
#pragma unroll
  for (int i = 0; i < 4; ++i) {
    int idx = i * 32 + lane;
    f4 v = src[idx];
    h4 hv;
#pragma unroll
    for (int j = 0; j < 4; ++j) hv[j] = (_Float16)v[j];
    *(h4*)(shh + (idx >> 3) * 32 + (idx & 7) * 4) = hv;
  }
  asm volatile("s_wait_dscnt 0" ::: "memory");
  v16h hA = load_a_lds(shh, lo, hi, 32);

  // layer 1: relu(h @ W1.T + b1) -> LDS r1 [16][256] f16
  for (int f = 0; f < 16; ++f) {
    v8f acc = wmma_f16(hA, load_b_frag(W1T + f * 512, lane),
                       splat8(b1[f * 16 + lo]));
#pragma unroll
    for (int v = 0; v < 8; ++v) {
      float x = acc[v];
      x = x > 0.0f ? x : 0.0f;
      shh[(v + 8 * hi) * 256 + f * 16 + lo] = (_Float16)x;
    }
  }
  asm volatile("s_wait_dscnt 0" ::: "memory");

  // cache all 8 K-chunk A fragments of r1
  v16h aF[8];
#pragma unroll
  for (int kc = 0; kc < 8; ++kc) aF[kc] = load_a_lds(shh + kc * 32, lo, hi, 256);

  // layer 2 + fused W3 dot: partial[v] = sum_c relu(r2)[row,c] * W3[c]
  float partial[8];
#pragma unroll
  for (int v = 0; v < 8; ++v) partial[v] = 0.0f;

  for (int f2 = 0; f2 < 16; ++f2) {
    v8f acc = splat8(b2[f2 * 16 + lo]);
#pragma unroll
    for (int kc = 0; kc < 8; ++kc)
      acc = wmma_f16(aF[kc], load_b_frag(W2T + (kc * 16 + f2) * 512, lane), acc);
    float w3c = W3[f2 * 16 + lo];
#pragma unroll
    for (int v = 0; v < 8; ++v) {
      float x = acc[v];
      x = x > 0.0f ? x : 0.0f;
      partial[v] += x * w3c;
    }
  }

  // cross-lane reduce: row r = v + 8*hi; lanes sharing hi cover all 256 cols
  float* red = (float*)shh;
#pragma unroll
  for (int v = 0; v < 8; ++v) red[v * 32 + lane] = partial[v];
  asm volatile("s_wait_dscnt 0" ::: "memory");
  if (lane < 16) {
    int r = lane;
    int vv = r & 7, lb = (r < 8) ? 0 : 16;
    float s = b3[0];
#pragma unroll
    for (int j = 0; j < 16; ++j) s += red[vv * 32 + lb + j];
    out[row0 + r] = s;
  }
}

// ---------------- host launch ----------------
extern "C" void kernel_launch(void* const* d_in, const int* in_sizes, int n_in,
                              void* d_out, int out_size, void* d_ws,
                              size_t ws_size, hipStream_t stream) {
  (void)in_sizes; (void)n_in; (void)out_size; (void)ws_size;

  const float* traffic        = (const float*)d_in[0];
  const float* packets        = (const float*)d_in[1];
  const float* tdp            = (const float*)d_in[2];
  const float* capacity       = (const float*)d_in[3];
  const int*   link_to_path   = (const int*)d_in[4];
  // d_in[5] path_ids, d_in[6] sequence_path: structure is implicit (p*PLEN+t)
  const int*   path_to_link   = (const int*)d_in[7];
  const int*   sequence_links = (const int*)d_in[8];
  // d_in[9] n_links, d_in[10] n_paths: static constants in reference
  const float* Wi_p = (const float*)d_in[11];
  const float* Wh_p = (const float*)d_in[12];
  const float* bi_p = (const float*)d_in[13];
  const float* bh_p = (const float*)d_in[14];
  const float* Wi_l = (const float*)d_in[15];
  const float* Wh_l = (const float*)d_in[16];
  const float* bi_l = (const float*)d_in[17];
  const float* bh_l = (const float*)d_in[18];
  const float* W1   = (const float*)d_in[19];
  const float* b1   = (const float*)d_in[20];
  const float* W2   = (const float*)d_in[21];
  const float* b2   = (const float*)d_in[22];
  const float* W3   = (const float*)d_in[23];
  const float* b3   = (const float*)d_in[24];

  char* ws = (char*)d_ws;
  float* path_state = (float*)(ws);                 // 50000*32*4 = 6,400,000 B
  float* link_state = (float*)(ws + 6400000);       // 10000*32*4 = 1,280,000 B
  float* path_sum   = (float*)(ws + 7680000);       // 10000*32*4 = 1,280,000 B
  _Float16* WiT_p = (_Float16*)(ws + 8960000);      // 6 frags * 512 halfs
  _Float16* WhT_p = WiT_p + 6 * 512;
  _Float16* WiT_l = WhT_p + 6 * 512;
  _Float16* WhT_l = WiT_l + 6 * 512;
  _Float16* W1T   = WhT_l + 6 * 512;                // 16 frags
  _Float16* W2T   = W1T + 16 * 512;                 // 128 frags

  init_states_kernel<<<6250, 256, 0, stream>>>(traffic, packets, tdp, capacity,
                                               path_state, link_state);
  pack_weight_kernel<<<12, 256, 0, stream>>>(Wi_p, WiT_p, 6, 1, 32);
  pack_weight_kernel<<<12, 256, 0, stream>>>(Wh_p, WhT_p, 6, 1, 32);
  pack_weight_kernel<<<12, 256, 0, stream>>>(Wi_l, WiT_l, 6, 1, 32);
  pack_weight_kernel<<<12, 256, 0, stream>>>(Wh_l, WhT_l, 6, 1, 32);
  pack_weight_kernel<<<32, 256, 0, stream>>>(W1, W1T, 16, 1, 32);
  pack_weight_kernel<<<256, 256, 0, stream>>>(W2, W2T, 16, 8, 256);

  const int pathBlocks = (N_PATHS / 16 + 3) / 4;  // 782
  const int linkBlocks = (N_LINKS / 16 + 3) / 4;  // 157
  for (int it = 0; it < T_ITERS; ++it) {
    path_gru_kernel<<<pathBlocks, 128, 0, stream>>>(
        link_state, path_state, link_to_path, WiT_p, WhT_p, bi_p, bh_p);
    zero_kernel<<<(N_LINKS * 32 + 255) / 256, 256, 0, stream>>>(path_sum,
                                                                N_LINKS * 32);
    scatter_add_kernel<<<(EDG + 255) / 256, 256, 0, stream>>>(
        path_state, path_to_link, sequence_links, path_sum);
    link_gru_kernel<<<linkBlocks, 128, 0, stream>>>(link_state, path_sum, WiT_l,
                                                    WhT_l, bi_l, bh_l);
  }
  readout_kernel<<<pathBlocks, 128, 0, stream>>>(path_state, W1T, b1, W2T, b2,
                                                 W3, b3, (float*)d_out);
}
